// PegasusXModel_71107478552597
// MI455X (gfx1250) — matrix operand order repaired and checked by reading, not compiled, not run on hardware
//
#include <hip/hip_runtime.h>
#include <hip/hip_bf16.h>

typedef _Float16 half_t;
typedef __attribute__((ext_vector_type(16))) _Float16 v16h;
typedef __attribute__((ext_vector_type(8)))  float    v8f;
typedef __attribute__((ext_vector_type(4)))  unsigned int u32x4;
typedef __attribute__((ext_vector_type(8)))  int      i32x8;
typedef __attribute__((ext_vector_type(4)))  int      i32x4;

// Problem constants
#define BB 2
#define SS 8192
#define DD 1024
#define HH 16
#define DKK 64
#define BSS 512
#define NN 16
#define GG 128
#define QSCALE 0.125f   // 1/sqrt(64)

#if __has_builtin(__builtin_amdgcn_tensor_load_to_lds)
#define HAVE_TDM 1
#else
#define HAVE_TDM 0
#endif

union FragU {
    v16h v;
    unsigned int u[8];
    half_t h[16];
};
union Q16 {
    uint4 q;
    half_t h[8];
};

__device__ __forceinline__ float hmax16(float v) {
    v = fmaxf(v, __shfl_xor(v, 1, 32));
    v = fmaxf(v, __shfl_xor(v, 2, 32));
    v = fmaxf(v, __shfl_xor(v, 4, 32));
    v = fmaxf(v, __shfl_xor(v, 8, 32));
    return v;
}
__device__ __forceinline__ float hsum16(float v) {
    v += __shfl_xor(v, 1, 32);
    v += __shfl_xor(v, 2, 32);
    v += __shfl_xor(v, 4, 32);
    v += __shfl_xor(v, 8, 32);
    return v;
}

#if HAVE_TDM
// ---------------------------------------------------------------------------
// Tensor Data Mover: 2D tile (tile_h rows x tile_w f16 elems, row stride
// stride_elems) from global 'gsrc' into LDS at byte offset 'lds_off', with
// LDS padding pad codes per D# spec (interval: 2^(c+1) DWORDs, amount: c+1).
// D# layout per CDNA5 ISA 8.3/8.4 (group0 128b, group1 256b, groups2/3 zero).
// ---------------------------------------------------------------------------
__device__ __forceinline__ void tdm_load_2d(const void* gsrc, unsigned lds_off,
                                            unsigned tile_w, unsigned tile_h,
                                            unsigned stride_elems,
                                            unsigned tensor_w, unsigned tensor_h,
                                            unsigned pad_int_code,
                                            unsigned pad_amt_code) {
    unsigned long long ga = (unsigned long long)(uintptr_t)gsrc;
    u32x4 g0;
    g0[0] = 1u;                                          // count=1 (valid user D#)
    g0[1] = lds_off;                                     // lds_addr
    g0[2] = (unsigned)ga;                                // global_addr[31:0]
    g0[3] = (unsigned)((ga >> 32) & 0x01FFFFFFull) | (2u << 30);  // addr[56:32] | type=2

    i32x8 g1;
    g1[0] = (int)((1u << 16)                             // data_size = 1 -> 2 bytes
                | (1u << 20)                             // pad_enable
                | (pad_int_code << 22)                   // pad_interval
                | (pad_amt_code << 25));                 // pad_amount
    g1[1] = (int)((tensor_w & 0xFFFFu) << 16);           // tensor_dim0[15:0]
    g1[2] = (int)((tensor_w >> 16) | ((tensor_h & 0xFFFFu) << 16)); // dim0 hi | dim1 lo
    g1[3] = (int)((tensor_h >> 16) | (tile_w << 16));    // dim1 hi | tile_dim0
    g1[4] = (int)(tile_h & 0xFFFFu);                     // tile_dim1 (tile_dim2=0)
    g1[5] = (int)stride_elems;                           // tensor_dim0_stride[31:0]
    g1[6] = 0;                                           // stride0 hi | stride1 lo
    g1[7] = 0;

    i32x4 z4 = {0, 0, 0, 0};
#if defined(__clang_major__) && (__clang_major__ >= 23)
    i32x8 z8 = {0, 0, 0, 0, 0, 0, 0, 0};
    __builtin_amdgcn_tensor_load_to_lds(g0, g1, z4, z4, z8, 0);
#else
    __builtin_amdgcn_tensor_load_to_lds(g0, g1, z4, z4, 0);
#endif
}
#endif // HAVE_TDM

// ---------------------------------------------------------------------------
// f32 -> f16 conversion
// ---------------------------------------------------------------------------
__global__ void cvt_f16_kernel(const float* __restrict__ src,
                               half_t* __restrict__ dst, long n) {
    long i = (long)blockIdx.x * blockDim.x + threadIdx.x;
    if (i < n) dst[i] = (half_t)src[i];
}

// ---------------------------------------------------------------------------
// Tiled WMMA GEMM:  C[M,Nd] = (A[M,K] * W[Nd,K]^T + bias) * scale
// 256 threads = 8 waves; 128x128 tile; K-step 32; wave computes 32x64.
// Tile staging by TDM (wave0 issues, double-buffered, pipelined one K-step
// ahead; s_wait_tensorcnt 2 => previous stage's two DMAs complete in order).
// ---------------------------------------------------------------------------
template <bool F16OUT>
__global__ __launch_bounds__(256)
void gemm_xwT_kernel(const half_t* __restrict__ A,
                     const half_t* __restrict__ W,
                     const float*  __restrict__ bias,
                     void* __restrict__ Cout,
                     int M, int Nd, int K, float scale) {
    __shared__ half_t sA[2][128][40];   // row stride 80B (16B aligned), pad 8 halves
    __shared__ half_t sW[2][128][40];

    const int tid  = threadIdx.x;
    const int wave = tid >> 5;
    const int lane = tid & 31;
    const int hs   = lane >> 4;
    const int lr   = lane & 15;

    const int rowBase = blockIdx.x * 128;
    const int colBase = blockIdx.y * 128;
    const int waveM   = wave & 3;
    const int waveN   = wave >> 2;

    v8f acc[2][4];
    #pragma unroll
    for (int i = 0; i < 2; ++i)
        #pragma unroll
        for (int j = 0; j < 4; ++j) {
            v8f z = {0, 0, 0, 0, 0, 0, 0, 0};
            acc[i][j] = z;
        }

    const int nk = K / 32;

#if HAVE_TDM
    // pad: interval code 3 -> every 16 DWORDs (32 halves); amount code 3 -> 4 DWORDs (8 halves)
    if (tid < 32) {
        tdm_load_2d(A + (size_t)rowBase * K, (unsigned)(uintptr_t)&sA[0][0][0],
                    32u, 128u, (unsigned)K, (unsigned)K, (unsigned)M, 3u, 3u);
        tdm_load_2d(W + (size_t)colBase * K, (unsigned)(uintptr_t)&sW[0][0][0],
                    32u, 128u, (unsigned)K, (unsigned)K, (unsigned)Nd, 3u, 3u);
    }
#endif

    for (int i = 0; i < nk; ++i) {
        const int bufI = i & 1;
#if HAVE_TDM
        if (tid < 32) {
            if (i + 1 < nk) {
                int nb = (i + 1) & 1;
                int k0 = (i + 1) * 32;
                tdm_load_2d(A + (size_t)rowBase * K + k0, (unsigned)(uintptr_t)&sA[nb][0][0],
                            32u, 128u, (unsigned)K, (unsigned)K, (unsigned)M, 3u, 3u);
                tdm_load_2d(W + (size_t)colBase * K + k0, (unsigned)(uintptr_t)&sW[nb][0][0],
                            32u, 128u, (unsigned)K, (unsigned)K, (unsigned)Nd, 3u, 3u);
                __builtin_amdgcn_s_wait_tensorcnt((short)2);
            } else {
                __builtin_amdgcn_s_wait_tensorcnt((short)0);
            }
        }
        __syncthreads();
#else
        // fallback: cooperative vector-load staging
        {
            int k0 = i * 32;
            #pragma unroll
            for (int it = 0; it < 2; ++it) {
                int idx = tid + it * 256;
                int row = idx >> 2;
                int seg = idx & 3;
                *(uint4*)&sA[bufI][row][seg * 8] =
                    *(const uint4*)(A + (size_t)(rowBase + row) * K + k0 + seg * 8);
                *(uint4*)&sW[bufI][row][seg * 8] =
                    *(const uint4*)(W + (size_t)(colBase + row) * K + k0 + seg * 8);
            }
        }
        __syncthreads();
#endif

        // A fragments (16x32, documented CDNA5 layout)
        v16h af[2];
        #pragma unroll
        for (int mg = 0; mg < 2; ++mg) {
            FragU f;
            int m  = waveM * 32 + mg * 16 + lr;
            int kb = hs * 8;
            #pragma unroll
            for (int v = 0; v < 4; ++v) f.u[v]     = *(const unsigned int*)&sA[bufI][m][kb + 2 * v];
            #pragma unroll
            for (int v = 0; v < 4; ++v) f.u[4 + v] = *(const unsigned int*)&sA[bufI][m][16 + kb + 2 * v];
            af[mg] = f.v;
        }
        // B fragments (32x16 = W^T tile)
        v16h bf[4];
        #pragma unroll
        for (int ng = 0; ng < 4; ++ng) {
            FragU f;
            int n  = waveN * 64 + ng * 16 + lr;
            int kb = hs * 16;
            #pragma unroll
            for (int j = 0; j < 8; ++j) f.u[j] = *(const unsigned int*)&sW[bufI][n][(j >> 2) * 16 + hs * 8 + 2 * (j & 3)];
            bf[ng] = f.v;
        }

        #pragma unroll
        for (int mg = 0; mg < 2; ++mg)
            #pragma unroll
            for (int ng = 0; ng < 4; ++ng)
                acc[mg][ng] = __builtin_amdgcn_wmma_f32_16x16x32_f16(
                    false, af[mg], false, bf[ng], (short)0, acc[mg][ng], false, false);
        __syncthreads();
    }

    // epilogue
    #pragma unroll
    for (int mg = 0; mg < 2; ++mg) {
        #pragma unroll
        for (int ng = 0; ng < 4; ++ng) {
            int nIdx = colBase + waveN * 64 + ng * 16 + lr;
            float bv = bias[nIdx];
            #pragma unroll
            for (int r = 0; r < 8; ++r) {
                int mIdx  = rowBase + waveM * 32 + mg * 16 + hs * 8 + r;
                float val = (acc[mg][ng][r] + bv) * scale;
                if (F16OUT)
                    ((half_t*)Cout)[(size_t)mIdx * Nd + nIdx] = (half_t)val;
                else
                    ((float*)Cout)[(size_t)mIdx * Nd + nIdx] = val;
            }
        }
    }
}

// ---------------------------------------------------------------------------
// Flash-style attention. 128 threads = 4 waves; each wave owns 16 queries;
// 32-key chunks with online softmax. K chunk staged row-major in LDS, V chunk
// staged TRANSPOSED in LDS so all WMMA B-fragments are aligned 4B LDS loads
// (no scalar half packing). All 4 waves share the staged chunk (4x less
// global traffic).
//   is_global == 0: keys = G global + 512 block-local.
//   is_global == 1: keys = G global + all S local; logits re-scaled by QSCALE.
// ---------------------------------------------------------------------------
__global__ __launch_bounds__(128)
void attn_kernel(const half_t* __restrict__ qsrc,
                 const half_t* __restrict__ gkh, const half_t* __restrict__ gvh,
                 const half_t* __restrict__ lkh, const half_t* __restrict__ lvh,
                 const float*  __restrict__ mask,
                 half_t* __restrict__ outp, int is_global) {
    __shared__ half_t sK[32][72];       // 32 keys x 64 dims (+8 pad)
    __shared__ half_t sVt[64][40];      // transposed: 64 dims x 32 keys (+8 pad)
    __shared__ half_t sP[4][16][34];    // per-wave P bounce (C-layout -> A-layout)

    const int tid  = threadIdx.x;
    const int bh   = blockIdx.z;
    const int b    = bh / HH;
    const int h    = bh % HH;
    const int wave = tid >> 5;
    const int lane = tid & 31;
    const int hs   = lane >> 4;
    const int lr   = lane & 15;

    size_t qRow0, lkeyRowBase;
    const float* maskBase;
    int   nLocal;
    float extra;
    if (is_global) {
        qRow0       = (size_t)b * GG + blockIdx.x * 64 + wave * 16;
        lkeyRowBase = (size_t)b * SS;
        maskBase    = mask + (size_t)b * SS;
        nLocal      = SS;
        extra       = QSCALE;          // logits scaled by 1/sqrt(dk) AGAIN
    } else {
        int blk     = blockIdx.y;
        qRow0       = (size_t)b * SS + blk * BSS + blockIdx.x * 64 + wave * 16;
        lkeyRowBase = (size_t)b * SS + blk * BSS;
        maskBase    = mask + (size_t)b * SS + blk * BSS;
        nLocal      = BSS;
        extra       = 1.0f;
    }

    // Q fragments (A-layout, K split 0..31 / 32..63), loaded once
    const half_t* qbase = qsrc + qRow0 * DD + h * DKK;
    v16h qa[2];
    #pragma unroll
    for (int kk = 0; kk < 2; ++kk) {
        FragU f;
        const half_t* qr = qbase + (size_t)lr * DD + kk * 32;
        int kb = hs * 8;
        #pragma unroll
        for (int v = 0; v < 4; ++v) f.u[v]     = *(const unsigned int*)(qr + kb + 2 * v);
        #pragma unroll
        for (int v = 0; v < 4; ++v) f.u[4 + v] = *(const unsigned int*)(qr + 16 + kb + 2 * v);
        qa[kk] = f.v;
    }

    v8f o[4];
    #pragma unroll
    for (int t = 0; t < 4; ++t) { v8f z = {0, 0, 0, 0, 0, 0, 0, 0}; o[t] = z; }
    float runmax[8], runsum[8];
    #pragma unroll
    for (int r = 0; r < 8; ++r) { runmax[r] = -3.0e38f; runsum[r] = 0.0f; }

    const int nch = (GG + nLocal) / 32;
    for (int c = 0; c < nch; ++c) {
        const half_t *kr0, *vr0;
        const float* mrow = nullptr;
        if (c < GG / 32) {
            size_t row0 = (size_t)b * GG + c * 32;
            kr0 = gkh + row0 * DD + h * DKK;
            vr0 = gvh + row0 * DD + h * DKK;
        } else {
            int    lk0  = (c - GG / 32) * 32;
            size_t row0 = lkeyRowBase + lk0;
            kr0  = lkh + row0 * DD + h * DKK;
            vr0  = lvh + row0 * DD + h * DKK;
            mrow = maskBase + lk0;
        }

        __syncthreads();   // previous chunk's LDS reads complete before overwrite
        // stage K (row-major) + V (transposed): 256 uint4 each, 2 per thread
        #pragma unroll
        for (int it = 0; it < 2; ++it) {
            int idx = tid + it * 128;     // 0..255
            int row = idx >> 3;           // key 0..31
            int seg = idx & 7;            // 8-half segment
            *(uint4*)&sK[row][seg * 8] = *(const uint4*)(kr0 + (size_t)row * DD + seg * 8);
            Q16 d;
            d.q = *(const uint4*)(vr0 + (size_t)row * DD + seg * 8);
            #pragma unroll
            for (int e = 0; e < 8; ++e) sVt[seg * 8 + e][row] = d.h[e];
        }
        __syncthreads();

        // ---- scores: two 16x16 tiles (keys 0..15, 16..31 of chunk) ----
        v8f s[2];
        #pragma unroll
        for (int t = 0; t < 2; ++t) {
            int krow = t * 16 + lr;
            FragU kb0, kb1;
            #pragma unroll
            for (int j = 0; j < 8; ++j) {
                kb0.u[j] = *(const unsigned int*)&sK[krow][(j >> 2) * 16 + hs * 8 + 2 * (j & 3)];
                kb1.u[j] = *(const unsigned int*)&sK[krow][32 + (j >> 2) * 16 + hs * 8 + 2 * (j & 3)];
            }
            v8f z = {0, 0, 0, 0, 0, 0, 0, 0};
            v8f tmp = __builtin_amdgcn_wmma_f32_16x16x32_f16(
                false, qa[0], false, kb0.v, (short)0, z, false, false);
            s[t] = __builtin_amdgcn_wmma_f32_16x16x32_f16(
                false, qa[1], false, kb1.v, (short)0, tmp, false, false);
            float mv = mrow ? mrow[t * 16 + lr] : 0.0f;
            #pragma unroll
            for (int r = 0; r < 8; ++r) s[t][r] = s[t][r] * extra + mv;
        }

        // ---- online softmax (rows live in 16-lane halves) ----
        #pragma unroll
        for (int r = 0; r < 8; ++r) {
            float cm = hmax16(fmaxf(s[0][r], s[1][r]));
            float nm = fmaxf(runmax[r], cm);
            float corr = __expf(runmax[r] - nm);
            runmax[r] = nm;
            runsum[r] *= corr;
            #pragma unroll
            for (int t = 0; t < 4; ++t) o[t][r] *= corr;
            s[0][r] = __expf(s[0][r] - nm);
            s[1][r] = __expf(s[1][r] - nm);
            runsum[r] += hsum16(s[0][r] + s[1][r]);
        }

        // ---- P: C-layout -> A-layout via per-wave LDS (intra-wave DS order) --
        #pragma unroll
        for (int t = 0; t < 2; ++t)
            #pragma unroll
            for (int r = 0; r < 8; ++r)
                sP[wave][hs * 8 + r][t * 16 + lr] = (half_t)s[t][r];

        FragU pf;
        {
            int kb = hs * 8;
            #pragma unroll
            for (int v = 0; v < 4; ++v) pf.u[v]     = *(const unsigned int*)&sP[wave][lr][kb + 2 * v];
            #pragma unroll
            for (int v = 0; v < 4; ++v) pf.u[4 + v] = *(const unsigned int*)&sP[wave][lr][16 + kb + 2 * v];
        }

        // ---- O += P(16x32) x V(32x64) : 4 WMMAs, B-frags from sVt ----
        #pragma unroll
        for (int t = 0; t < 4; ++t) {
            FragU vf;
            #pragma unroll
            for (int j = 0; j < 8; ++j)
                vf.u[j] = *(const unsigned int*)&sVt[t * 16 + lr][(j >> 2) * 16 + hs * 8 + 2 * (j & 3)];
            o[t] = __builtin_amdgcn_wmma_f32_16x16x32_f16(
                false, pf.v, false, vf.v, (short)0, o[t], false, false);
        }
    }

    // ---- normalize, store merged-head layout [row, D] ----
    half_t* obase = outp + qRow0 * DD + h * DKK;
    #pragma unroll
    for (int t = 0; t < 4; ++t) {
        #pragma unroll
        for (int r = 0; r < 8; ++r) {
            int m = hs * 8 + r;
            float val = o[t][r] / runsum[r];
            obase[(size_t)m * DD + t * 16 + lr] = (half_t)val;
        }
    }
}

// ---------------------------------------------------------------------------
// Host orchestration
// ---------------------------------------------------------------------------
extern "C" void kernel_launch(void* const* d_in, const int* in_sizes, int n_in,
                              void* d_out, int out_size, void* d_ws, size_t ws_size,
                              hipStream_t stream) {
    (void)in_sizes; (void)n_in; (void)out_size; (void)ws_size;

    const float* Q    = (const float*)d_in[0];
    const float* K    = (const float*)d_in[1];
    const float* V    = (const float*)d_in[2];
    const float* Gt   = (const float*)d_in[3];
    const float* mask = (const float*)d_in[4];
    const float* Wq   = (const float*)d_in[5];
    const float* bq   = (const float*)d_in[6];
    const float* Wk   = (const float*)d_in[7];
    const float* bk   = (const float*)d_in[8];
    const float* Wv   = (const float*)d_in[9];
    const float* bv   = (const float*)d_in[10];
    const float* Wo   = (const float*)d_in[11];
    const float* bo   = (const float*)d_in[12];

    const size_t nSD = (size_t)BB * SS * DD;
    const size_t nGD = (size_t)BB * GG * DD;
    const size_t nW  = (size_t)DD * DD;

    char* wsp = (char*)d_ws;
    size_t off = 0;
    auto alloc_h = [&](size_t nelem) -> half_t* {
        half_t* p = (half_t*)(wsp + off);
        off = (off + nelem * sizeof(half_t) + 255) & ~(size_t)255;
        return p;
    };
    half_t* Qh  = alloc_h(nSD);
    half_t* Kh  = alloc_h(nSD);
    half_t* Vh  = alloc_h(nSD);
    half_t* Gh  = alloc_h(nGD);
    half_t* Wqh = alloc_h(nW);
    half_t* Wkh = alloc_h(nW);
    half_t* Wvh = alloc_h(nW);
    half_t* Woh = alloc_h(nW);
    half_t* lqh = alloc_h(nSD);
    half_t* lkh = alloc_h(nSD);
    half_t* lvh = alloc_h(nSD);
    half_t* gqh = alloc_h(nGD);
    half_t* gkh = alloc_h(nGD);
    half_t* gvh = alloc_h(nGD);
    half_t* loh = alloc_h(nSD);
    half_t* goh = alloc_h(nGD);

    auto cvt = [&](const float* s, half_t* d, size_t n) {
        cvt_f16_kernel<<<dim3((unsigned)((n + 255) / 256)), 256, 0, stream>>>(s, d, (long)n);
    };
    cvt(Q, Qh, nSD);  cvt(K, Kh, nSD);  cvt(V, Vh, nSD);  cvt(Gt, Gh, nGD);
    cvt(Wq, Wqh, nW); cvt(Wk, Wkh, nW); cvt(Wv, Wvh, nW); cvt(Wo, Woh, nW);

    auto gemm16 = [&](const half_t* A, const half_t* W, const float* bias,
                      half_t* C, int M, float scale) {
        dim3 g(M / 128, DD / 128);
        gemm_xwT_kernel<true><<<g, 256, 0, stream>>>(A, W, bias, (void*)C, M, DD, DD, scale);
    };
    auto gemm32 = [&](const half_t* A, const half_t* W, const float* bias,
                      float* C, int M) {
        dim3 g(M / 128, DD / 128);
        gemm_xwT_kernel<false><<<g, 256, 0, stream>>>(A, W, bias, (void*)C, M, DD, DD, 1.0f);
    };

    const int Ml = BB * SS;
    const int Mg = BB * GG;

    gemm16(Qh, Wqh, bq, lqh, Ml, QSCALE);
    gemm16(Kh, Wkh, bk, lkh, Ml, 1.0f);
    gemm16(Vh, Wvh, bv, lvh, Ml, 1.0f);
    gemm16(Gh, Wqh, bq, gqh, Mg, QSCALE);
    gemm16(Gh, Wkh, bk, gkh, Mg, 1.0f);
    gemm16(Gh, Wvh, bv, gvh, Mg, 1.0f);

    {
        dim3 g(BSS / 64, NN, BB * HH);
        attn_kernel<<<g, 128, 0, stream>>>(lqh, gkh, gvh, lkh, lvh, mask, loh, 0);
    }
    {
        dim3 g(GG / 64, 1, BB * HH);
        attn_kernel<<<g, 128, 0, stream>>>(gqh, gkh, gvh, lkh, lvh, mask, goh, 1);
    }

    float* out_local  = (float*)d_out;
    float* out_global = (float*)d_out + nSD;
    gemm32(loh, Woh, bo, out_local, Ml);
    gemm32(goh, Woh, bo, out_global, Mg);
}
